// GPSLayer_9225589751900
// MI455X (gfx1250) — compile-verified, matrix operand order checked by reading, not run on hardware
//
#include <hip/hip_runtime.h>
#include <cstdint>
#include <cstddef>

// ---------------------------------------------------------------------------
// GPS layer (GATv2 + per-graph MHA + gated combine + FFN) for MI455X/gfx1250.
// All dense math runs on bf16 WMMA (v_wmma_f32_16x16x32_bf16, f32 accum).
// A-tiles of the GEMMs are fetched by the Tensor Data Mover
// (tensor_load_to_lds + s_wait_tensorcnt); B panels staged transposed in LDS.
// ---------------------------------------------------------------------------

#define USE_TDM 1

static const int Nn   = 32768;
static const int Ee   = 524288;
static const int Hh   = 8;
static const int Gg   = 64;

typedef __bf16 bf16;
typedef bf16         v16bf __attribute__((ext_vector_type(16)));
typedef float        v8f   __attribute__((ext_vector_type(8)));
typedef unsigned int u32x4 __attribute__((ext_vector_type(4)));
typedef int          i32x4 __attribute__((ext_vector_type(4)));
typedef int          i32x8 __attribute__((ext_vector_type(8)));

union Frag  { v16bf v; u32x4 q[2]; };   // trivially-constructible (no HIP uint4)
union B16x8 { u32x4 q; bf16 h[8]; };

// order-preserving float<->uint encode for atomicMax-based segment max
__device__ __forceinline__ unsigned encf(float f) {
    unsigned u = __float_as_uint(f);
    return (u & 0x80000000u) ? ~u : (u | 0x80000000u);
}
__device__ __forceinline__ float decf(unsigned u) {
    unsigned b = (u & 0x80000000u) ? (u ^ 0x80000000u) : ~u;
    return __uint_as_float(b);
}

// ---------------------------------------------------------------------------
// Weight packing: everything to bf16, fused B panel [W_l | W_r | in_proj^T]
// ---------------------------------------------------------------------------
__global__ __launch_bounds__(256) void k_pack(
    const float* Wl, const float* bl, const float* Wr, const float* br,
    const float* inw, const float* inb, const float* opw,
    const float* W1, const float* W2,
    bf16* Wcat, float* bcat, bf16* opT, bf16* W1b, bf16* W2b)
{
    const int S1 = 256 * 1280, S2 = 1280, S3 = 256 * 256, S4 = 256 * 1024, S5 = 1024 * 256;
    int i = blockIdx.x * 256 + threadIdx.x;
    if (i < S1) {
        int k = i / 1280, j = i % 1280;
        float v = (j < 256) ? Wl[k * 256 + j]
                : (j < 512) ? Wr[k * 256 + (j - 256)]
                            : inw[(j - 512) * 256 + k];   // qkv = x @ in_proj_w^T
        Wcat[i] = (bf16)v;
        return;
    }
    i -= S1;
    if (i < S2) { bcat[i] = (i < 256) ? bl[i] : (i < 512) ? br[i - 256] : inb[i - 512]; return; }
    i -= S2;
    if (i < S3) { int k = i / 256, j = i % 256; opT[i] = (bf16)opw[j * 256 + k]; return; }
    i -= S3;
    if (i < S4) { W1b[i] = (bf16)W1[i]; return; }
    i -= S4;
    if (i < S5) { W2b[i] = (bf16)W2[i]; return; }
}

__global__ __launch_bounds__(256) void k_cvt(const float* x, bf16* xb, int n) {
    int i = blockIdx.x * 256 + threadIdx.x;
    if (i < n) xb[i] = (bf16)x[i];
}

// ---------------------------------------------------------------------------
// Generic bf16 WMMA GEMM:  C[M,Nc] = A[M,K](bf16) * B[K,Nc](bf16) + bias
// block = 256 threads (8 waves), tile 128x128, k-step 32.
// wave (wm in 0..3, wn in 0..1) owns a 32x64 sub-tile = 2x4 WMMA accumulators.
// LDS row stride 40 bf16 (80B): 16B aligned b128 frag loads, conflict-free.
// A tile is DMA'd by the TDM with pad_interval=16DW / pad_amount=4DW to
// produce the 40-element stride directly.
// ---------------------------------------------------------------------------
__global__ __launch_bounds__(256) void k_gemm(
    const bf16* __restrict__ A, const bf16* __restrict__ B,
    const float* __restrict__ bias, float* __restrict__ Cf, bf16* __restrict__ Cb,
    int M, int Nc, int K, int ldb, int ldc, int act)
{
    __shared__ bf16 Alds[128 * 40];
    __shared__ bf16 Btl[128 * 40];

    const int t = threadIdx.x;
    const int w = t >> 5, lane = t & 31;
    const int wm = w & 3, wn = w >> 2;
    const int m0 = blockIdx.y * 128, n0 = blockIdx.x * 128;
    const int lrow = lane & 15;
    const int kb   = (lane >> 4) << 3;   // K-chunk base: 0 or 8
    const int rlo  = (lane >> 4) << 3;   // C-layout row offset: 0 or 8

    v8f acc[2][4];
    #pragma unroll
    for (int s = 0; s < 2; ++s)
        #pragma unroll
        for (int f = 0; f < 4; ++f)
            #pragma unroll
            for (int r = 0; r < 8; ++r) acc[s][f][r] = 0.0f;

    for (int k0 = 0; k0 < K; k0 += 32) {
        __syncthreads();
#if USE_TDM
        if (t < 32) {  // wave 0 drives the Tensor Data Mover for the A tile
            unsigned lds_off = (unsigned)(size_t)(&Alds[0]);
            unsigned long long ga = (unsigned long long)(size_t)(A + (size_t)m0 * K + k0);
            u32x4 g0;
            g0[0] = 1u;                                   // count=1, user mode
            g0[1] = lds_off;                              // lds_addr
            g0[2] = (unsigned)(ga & 0xFFFFFFFFu);         // global_addr lo
            g0[3] = (unsigned)((ga >> 32) & 0x01FFFFFFu) | 0x80000000u; // hi | type=2
            i32x8 g1;
            g1[0] = (1 << 16) | (1 << 20) | (3 << 22) | (3 << 25); // 2B elems, pad 16DW->+4DW
            g1[1] = (int)(((unsigned)K & 0xFFFFu) << 16);          // tensor_dim0 lo
            g1[2] = (int)(((unsigned)K >> 16) | (((unsigned)M & 0xFFFFu) << 16));
            g1[3] = (int)(((unsigned)M >> 16) | (32u << 16));      // tile_dim0 = 32
            g1[4] = 128;                                           // tile_dim1 = 128
            g1[5] = K;                                             // dim0 stride lo
            g1[6] = 0; g1[7] = 0;
            i32x4 gz; gz[0] = 0; gz[1] = 0; gz[2] = 0; gz[3] = 0;
            i32x8 gz8;
            #pragma unroll
            for (int z = 0; z < 8; ++z) gz8[z] = 0;
            __builtin_amdgcn_tensor_load_to_lds(g0, g1, gz, gz, gz8, 0);
        }
#else
        {
            int rr = t >> 1, ch = (t & 1) << 4;
            const bf16* src = A + (size_t)(m0 + rr) * K + k0 + ch;
            *(u32x4*)&Alds[rr * 40 + ch]     = *(const u32x4*)src;
            *(u32x4*)&Alds[rr * 40 + ch + 8] = *(const u32x4*)(src + 8);
        }
#endif
        {   // stage B transposed: Btl[n][k]
            int kk = t >> 3, ng = (t & 7) << 4;
            const bf16* src = B + (size_t)(k0 + kk) * ldb + n0 + ng;
            if (k0 + 32 < K) __builtin_prefetch(src + (size_t)32 * ldb, 0, 1);
            B16x8 u0, u1;
            u0.q = *(const u32x4*)src;
            u1.q = *(const u32x4*)(src + 8);
            #pragma unroll
            for (int i = 0; i < 8; ++i) Btl[(ng + i) * 40 + kk]     = u0.h[i];
            #pragma unroll
            for (int i = 0; i < 8; ++i) Btl[(ng + 8 + i) * 40 + kk] = u1.h[i];
        }
#if USE_TDM
        if (t < 32) __builtin_amdgcn_s_wait_tensorcnt(0);
#endif
        __syncthreads();

        Frag a[2], b[4];
        #pragma unroll
        for (int s = 0; s < 2; ++s) {
            const bf16* p = &Alds[(wm * 32 + s * 16 + lrow) * 40 + kb];
            a[s].q[0] = *(const u32x4*)p;
            a[s].q[1] = *(const u32x4*)(p + 16);
        }
        #pragma unroll
        for (int f = 0; f < 4; ++f) {
            const bf16* p = &Btl[(wn * 64 + f * 16 + lrow) * 40 + kb];
            b[f].q[0] = *(const u32x4*)p;
            b[f].q[1] = *(const u32x4*)(p + 16);
        }
        #pragma unroll
        for (int s = 0; s < 2; ++s)
            #pragma unroll
            for (int f = 0; f < 4; ++f)
                acc[s][f] = __builtin_amdgcn_wmma_f32_16x16x32_bf16(
                    false, a[s].v, false, b[f].v, (short)0, acc[s][f], false, false);
    }

    #pragma unroll
    for (int f = 0; f < 4; ++f) {
        int col = n0 + wn * 64 + f * 16 + lrow;
        float bv = bias ? bias[col] : 0.0f;
        #pragma unroll
        for (int s = 0; s < 2; ++s) {
            #pragma unroll
            for (int r = 0; r < 8; ++r) {
                float v = acc[s][f][r] + bv;
                if (act == 1) v = 0.5f * v * (1.0f + erff(v * 0.70710678118f)); // exact GELU
                size_t row = (size_t)(m0 + wm * 32 + s * 16 + rlo + r);
                if (Cf) Cf[row * (size_t)ldc + col] = v;
                if (Cb) Cb[row * (size_t)ldc + col] = (bf16)v;
            }
        }
    }
}

// ---------------------------------------------------------------------------
// Flash attention, one block per (graph, head). q/k/v read from the fused
// out1 buffer (cols 512.. = q, 768.. = k, 1024.. = v, ldc=1280).
// Online softmax: row stats live in registers, reduced with shfl_xor over
// the 16-lane halves matching the WMMA C layout.
// ---------------------------------------------------------------------------
__global__ __launch_bounds__(256) void k_flash(const float* __restrict__ out1,
                                               bf16* __restrict__ attn)
{
    __shared__ bf16 Kt[128 * 40];    // [key][c], stride 40
    __shared__ bf16 Vt[32 * 136];    // [c][key], stride 136
    __shared__ bf16 Pl[128 * 136];   // per-wave 16-row P scratch

    const int gh = blockIdx.x;
    const int g = gh >> 3, h = gh & 7;
    const int t = threadIdx.x;
    const int w = t >> 5, lane = t & 31;
    const int lrow = lane & 15;
    const int kb  = (lane >> 4) << 3;
    const int rlo = (lane >> 4) << 3;
    const float rsc = 0.17677669529f;  // 1/sqrt(32)

    for (int qt = 0; qt < 4; ++qt) {
        Frag qf;  // Q fragment, pre-scaled by 1/sqrt(C)
        {
            int qrow = g * 512 + qt * 128 + w * 16 + lrow;
            const float* qp = out1 + (size_t)qrow * 1280 + 512 + h * 32 + kb;
            B16x8 u0, u1;
            #pragma unroll
            for (int j = 0; j < 8; ++j) { u0.h[j] = (bf16)(qp[j] * rsc); u1.h[j] = (bf16)(qp[16 + j] * rsc); }
            qf.q[0] = u0.q; qf.q[1] = u1.q;
        }
        float mrun[8], lrun[8];
        v8f O0, O1;
        #pragma unroll
        for (int r = 0; r < 8; ++r) { mrun[r] = -1e30f; lrun[r] = 0.0f; O0[r] = 0.0f; O1[r] = 0.0f; }

        for (int kc = 0; kc < 4; ++kc) {
            __syncthreads();
            {   // stage K chunk: Kt[key][c]
                int key = t >> 1, ch = (t & 1) << 4;
                const float* kp = out1 + (size_t)(g * 512 + kc * 128 + key) * 1280 + 768 + h * 32 + ch;
                #pragma unroll
                for (int j = 0; j < 16; ++j) Kt[key * 40 + ch + j] = (bf16)kp[j];
            }
            {   // stage V chunk transposed: Vt[c][key]
                int c = t >> 3, kg = (t & 7) << 4;
                const float* vp = out1 + (size_t)(g * 512 + kc * 128 + kg) * 1280 + 1024 + h * 32 + c;
                #pragma unroll
                for (int j = 0; j < 16; ++j) Vt[c * 136 + kg + j] = (bf16)vp[(size_t)j * 1280];
            }
            __syncthreads();

            v8f S[8];
            #pragma unroll
            for (int tt = 0; tt < 8; ++tt) {   // S = Q * K^T (16 x 128)
                Frag bk;
                const bf16* p = &Kt[(tt * 16 + lrow) * 40 + kb];
                bk.q[0] = *(const u32x4*)p; bk.q[1] = *(const u32x4*)(p + 16);
                v8f z;
                #pragma unroll
                for (int r = 0; r < 8; ++r) z[r] = 0.0f;
                S[tt] = __builtin_amdgcn_wmma_f32_16x16x32_bf16(
                    false, qf.v, false, bk.v, (short)0, z, false, false);
            }
            // online softmax update per row
            #pragma unroll
            for (int r = 0; r < 8; ++r) {
                float cm = S[0][r];
                #pragma unroll
                for (int tt = 1; tt < 8; ++tt) cm = fmaxf(cm, S[tt][r]);
                #pragma unroll
                for (int off = 8; off > 0; off >>= 1) cm = fmaxf(cm, __shfl_xor(cm, off, 16));
                float nm = fmaxf(mrun[r], cm);
                float fr = __expf(mrun[r] - nm);
                float ps = 0.0f;
                #pragma unroll
                for (int tt = 0; tt < 8; ++tt) { float p = __expf(S[tt][r] - nm); S[tt][r] = p; ps += p; }
                #pragma unroll
                for (int off = 8; off > 0; off >>= 1) ps += __shfl_xor(ps, off, 16);
                lrun[r] = lrun[r] * fr + ps;
                mrun[r] = nm;
                O0[r] *= fr; O1[r] *= fr;
            }
            // C-layout -> A-layout via per-wave LDS scratch
            #pragma unroll
            for (int tt = 0; tt < 8; ++tt)
                #pragma unroll
                for (int r = 0; r < 8; ++r)
                    Pl[(w * 16 + rlo + r) * 136 + tt * 16 + lrow] = (bf16)S[tt][r];
            // O += P * V
            #pragma unroll
            for (int kc2 = 0; kc2 < 4; ++kc2) {
                Frag pf, v0, v1;
                const bf16* pp = &Pl[(w * 16 + lrow) * 136 + kc2 * 32 + kb];
                pf.q[0] = *(const u32x4*)pp; pf.q[1] = *(const u32x4*)(pp + 16);
                const bf16* vv0 = &Vt[lrow * 136 + kc2 * 32 + kb];
                const bf16* vv1 = &Vt[(16 + lrow) * 136 + kc2 * 32 + kb];
                v0.q[0] = *(const u32x4*)vv0; v0.q[1] = *(const u32x4*)(vv0 + 16);
                v1.q[0] = *(const u32x4*)vv1; v1.q[1] = *(const u32x4*)(vv1 + 16);
                O0 = __builtin_amdgcn_wmma_f32_16x16x32_bf16(false, pf.v, false, v0.v, (short)0, O0, false, false);
                O1 = __builtin_amdgcn_wmma_f32_16x16x32_bf16(false, pf.v, false, v1.v, (short)0, O1, false, false);
            }
        }
        int rowb = g * 512 + qt * 128 + w * 16 + rlo;
        #pragma unroll
        for (int r = 0; r < 8; ++r) {
            float inv = 1.0f / lrun[r];
            attn[(size_t)(rowb + r) * 256 + h * 32 + lrow]      = (bf16)(O0[r] * inv);
            attn[(size_t)(rowb + r) * 256 + h * 32 + 16 + lrow] = (bf16)(O1[r] * inv);
        }
    }
}

// ---------------------------------------------------------------------------
// GATv2 edge path
// ---------------------------------------------------------------------------
__global__ __launch_bounds__(256) void k_edge_prep(const int* ei, const float* ea,
                                                   float* deg, float* easum)
{
    int e = blockIdx.x * 256 + threadIdx.x;
    if (e >= Ee) return;
    int dst = ei[Ee + e];
    atomicAdd(&deg[dst], 1.0f);
    atomicAdd(&easum[dst * 2 + 0], ea[e * 2 + 0]);
    atomicAdd(&easum[dst * 2 + 1], ea[e * 2 + 1]);
}

__global__ __launch_bounds__(256) void k_eamean(const float* deg, const float* easum, float* eam)
{
    int i = blockIdx.x * 256 + threadIdx.x;
    if (i >= Nn) return;
    float d = fmaxf(deg[i], 1.0f);
    eam[i * 2 + 0] = easum[i * 2 + 0] / d;
    eam[i * 2 + 1] = easum[i * 2 + 1] / d;
}

__global__ __launch_bounds__(256) void k_score(const int* ei, const float* ea, const float* eam,
                                               const float* out1, const float* We, const float* att,
                                               float* score, unsigned* smax)
{
    int eh = blockIdx.x * 256 + threadIdx.x;
    if (eh >= (Ee + Nn) * Hh) return;
    int e = eh >> 3, h = eh & 7;
    int src, dst; float a0, a1;
    if (e < Ee) { src = ei[e]; dst = ei[Ee + e]; a0 = ea[e * 2]; a1 = ea[e * 2 + 1]; }
    else        { src = dst = e - Ee; a0 = eam[src * 2]; a1 = eam[src * 2 + 1]; }
    const float* xl = out1 + (size_t)src * 1280 + h * 32;
    const float* xr = out1 + (size_t)dst * 1280 + 256 + h * 32;
    const float* w0 = We + h * 32;
    const float* w1 = We + 256 + h * 32;
    const float* at = att + h * 32;
    float s = 0.0f;
    #pragma unroll 8
    for (int c = 0; c < 32; ++c) {
        float v = xl[c] + xr[c] + a0 * w0[c] + a1 * w1[c];
        v = v > 0.0f ? v : 0.2f * v;            // leaky_relu(0.2)
        s += v * at[c];
    }
    score[eh] = s;
    atomicMax(&smax[dst * 8 + h], encf(s));
}

__global__ __launch_bounds__(256) void k_exps(const int* ei, float* score,
                                              const unsigned* smax, float* ssum)
{
    int eh = blockIdx.x * 256 + threadIdx.x;
    if (eh >= (Ee + Nn) * Hh) return;
    int e = eh >> 3, h = eh & 7;
    int dst = (e < Ee) ? ei[Ee + e] : (e - Ee);
    float ex = __expf(score[eh] - decf(smax[dst * 8 + h]));
    score[eh] = ex;
    atomicAdd(&ssum[dst * 8 + h], ex);
}

__global__ __launch_bounds__(256) void k_scatter(const int* ei, const float* out1,
                                                 const float* score, const float* ssum, float* gat)
{
    int eh = blockIdx.x * 256 + threadIdx.x;
    if (eh >= (Ee + Nn) * Hh) return;
    int e = eh >> 3, h = eh & 7;
    int src, dst;
    if (e < Ee) { src = ei[e]; dst = ei[Ee + e]; } else { src = dst = e - Ee; }
    float alpha = score[eh] / ssum[dst * 8 + h];
    const float* xl = out1 + (size_t)src * 1280 + h * 32;
    float* o = gat + (size_t)dst * 256 + h * 32;
    #pragma unroll 8
    for (int c = 0; c < 32; ++c) atomicAdd(&o[c], alpha * xl[c]);
}

// ---------------------------------------------------------------------------
// LayerNorm / gate / final
// ---------------------------------------------------------------------------
__device__ __forceinline__ float blockMean(float v, float* red) {
    red[threadIdx.x] = v;
    __syncthreads();
    for (int s = 128; s > 0; s >>= 1) {
        if ((int)threadIdx.x < s) red[threadIdx.x] += red[threadIdx.x + s];
        __syncthreads();
    }
    float m = red[0] * (1.0f / 256.0f);
    __syncthreads();
    return m;
}

__global__ __launch_bounds__(256) void k_combine(const float* gat, const float* bgat,
                                                 const float* mha,
                                                 const float* ga1, const float* bb1,
                                                 const float* ga2, const float* bb2,
                                                 const float* alpha,
                                                 float* comb, bf16* combb)
{
    __shared__ float red[256];
    size_t n = blockIdx.x; int t = threadIdx.x;
    float lv = gat[n * 256 + t] + bgat[t];
    float m1 = blockMean(lv, red);
    float d1 = lv - m1;
    float v1 = blockMean(d1 * d1, red);
    float l  = d1 * rsqrtf(v1 + 1e-5f) * ga1[t] + bb1[t];
    float gv = mha[n * 256 + t];
    float m2 = blockMean(gv, red);
    float d2 = gv - m2;
    float v2 = blockMean(d2 * d2, red);
    float gl = d2 * rsqrtf(v2 + 1e-5f) * ga2[t] + bb2[t];
    float a  = 1.0f / (1.0f + __expf(-alpha[0]));
    float c  = a * l + (1.0f - a) * gl;
    comb[n * 256 + t]  = c;
    combb[n * 256 + t] = (bf16)c;
}

__global__ __launch_bounds__(256) void k_final(const float* comb, const float* ffn,
                                               const float* ga3, const float* bb3, float* out)
{
    __shared__ float red[256];
    size_t n = blockIdx.x; int t = threadIdx.x;
    float v = comb[n * 256 + t] + ffn[n * 256 + t];
    float m = blockMean(v, red);
    float d = v - m;
    float vv = blockMean(d * d, red);
    out[n * 256 + t] = d * rsqrtf(vv + 1e-5f) * ga3[t] + bb3[t];
}

// ---------------------------------------------------------------------------
// Host launcher
// ---------------------------------------------------------------------------
extern "C" void kernel_launch(void* const* d_in, const int* in_sizes, int n_in,
                              void* d_out, int out_size, void* d_ws, size_t ws_size,
                              hipStream_t stream)
{
    const float* x        = (const float*)d_in[0];
    const float* eattr    = (const float*)d_in[1];
    const float* W_l      = (const float*)d_in[2];
    const float* b_l      = (const float*)d_in[3];
    const float* W_r      = (const float*)d_in[4];
    const float* b_r      = (const float*)d_in[5];
    const float* W_e      = (const float*)d_in[6];
    const float* att      = (const float*)d_in[7];
    const float* bgat     = (const float*)d_in[8];
    const float* in_w     = (const float*)d_in[9];
    const float* in_b     = (const float*)d_in[10];
    const float* op_w     = (const float*)d_in[11];
    const float* op_b     = (const float*)d_in[12];
    const float* alpha    = (const float*)d_in[13];
    const float* W1       = (const float*)d_in[14];
    const float* b1       = (const float*)d_in[15];
    const float* W2       = (const float*)d_in[16];
    const float* b2       = (const float*)d_in[17];
    const float* ga1      = (const float*)d_in[18];
    const float* bb1      = (const float*)d_in[19];
    const float* ga2      = (const float*)d_in[20];
    const float* bb2      = (const float*)d_in[21];
    const float* ga3      = (const float*)d_in[22];
    const float* bb3      = (const float*)d_in[23];
    const int*   eidx     = (const int*)d_in[24];

    char* base = (char*)d_ws;
    size_t off = 0;
    auto alloc = [&](size_t bytes) -> char* {
        char* p = base + off;
        off = (off + bytes + 255) & ~(size_t)255;
        return p;
    };
    bf16*  xb     = (bf16*) alloc((size_t)Nn * 256 * 2);
    bf16*  Wcat   = (bf16*) alloc((size_t)256 * 1280 * 2);
    float* bcat   = (float*)alloc((size_t)1280 * 4);
    float* out1   = (float*)alloc((size_t)Nn * 1280 * 4);     // [xl | xr | q | k | v]
    bf16*  opT    = (bf16*) alloc((size_t)256 * 256 * 2);
    bf16*  W1b    = (bf16*) alloc((size_t)256 * 1024 * 2);
    bf16*  W2b    = (bf16*) alloc((size_t)1024 * 256 * 2);
    bf16*  attnb  = (bf16*) alloc((size_t)Nn * 256 * 2);
    float* mha    = (float*)alloc((size_t)Nn * 256 * 4);
    float* deg    = (float*)alloc((size_t)Nn * 4);
    float* easum  = (float*)alloc((size_t)Nn * 2 * 4);
    float* eam    = (float*)alloc((size_t)Nn * 2 * 4);
    float* score  = (float*)alloc((size_t)(Ee + Nn) * 8 * 4);
    unsigned* smax= (unsigned*)alloc((size_t)Nn * 8 * 4);
    float* ssum   = (float*)alloc((size_t)Nn * 8 * 4);
    float* gat    = (float*)alloc((size_t)Nn * 256 * 4);
    float* comb   = (float*)alloc((size_t)Nn * 256 * 4);
    bf16*  combb  = (bf16*) alloc((size_t)Nn * 256 * 2);
    bf16*  hidb   = (bf16*) alloc((size_t)Nn * 1024 * 2);
    float* ffn    = (float*)alloc((size_t)Nn * 256 * 4);
    (void)ws_size; (void)in_sizes; (void)n_in; (void)out_size;

    (void)hipMemsetAsync(deg,   0, (size_t)Nn * 4, stream);
    (void)hipMemsetAsync(easum, 0, (size_t)Nn * 2 * 4, stream);
    (void)hipMemsetAsync(smax,  0, (size_t)Nn * 8 * 4, stream);   // enc(-NaN): below all reals
    (void)hipMemsetAsync(ssum,  0, (size_t)Nn * 8 * 4, stream);
    (void)hipMemsetAsync(gat,   0, (size_t)Nn * 256 * 4, stream);

    const dim3 blk(256);
    {
        int total = 256 * 1280 + 1280 + 256 * 256 + 256 * 1024 + 1024 * 256;
        k_pack<<<(total + 255) / 256, blk, 0, stream>>>(W_l, b_l, W_r, b_r, in_w, in_b,
                                                        op_w, W1, W2, Wcat, bcat, opT, W1b, W2b);
    }
    k_cvt<<<(Nn * 256) / 256, blk, 0, stream>>>(x, xb, Nn * 256);

    // fused projection GEMM: out1 = x @ [W_l | W_r | in_proj^T] + bias
    k_gemm<<<dim3(1280 / 128, Nn / 128), blk, 0, stream>>>(
        xb, Wcat, bcat, out1, (bf16*)nullptr, Nn, 1280, 256, 1280, 1280, 0);

    // GATv2 edge path
    k_edge_prep<<<(Ee + 255) / 256, blk, 0, stream>>>(eidx, eattr, deg, easum);
    k_eamean<<<Nn / 256, blk, 0, stream>>>(deg, easum, eam);
    {
        int tot = (Ee + Nn) * Hh;
        k_score  <<<(tot + 255) / 256, blk, 0, stream>>>(eidx, eattr, eam, out1, W_e, att, score, smax);
        k_exps   <<<(tot + 255) / 256, blk, 0, stream>>>(eidx, score, smax, ssum);
        k_scatter<<<(tot + 255) / 256, blk, 0, stream>>>(eidx, out1, score, ssum, gat);
    }

    // per-graph flash attention + out projection
    k_flash<<<Gg * Hh, blk, 0, stream>>>(out1, attnb);
    k_gemm<<<dim3(256 / 128, Nn / 128), blk, 0, stream>>>(
        attnb, opT, op_b, mha, (bf16*)nullptr, Nn, 256, 256, 256, 256, 0);

    // LN + gated combine
    k_combine<<<Nn, blk, 0, stream>>>(gat, bgat, mha, ga1, bb1, ga2, bb2, alpha, comb, combb);

    // FFN (GELU) + final residual LN
    k_gemm<<<dim3(1024 / 128, Nn / 128), blk, 0, stream>>>(
        combb, W1b, b1, (float*)nullptr, hidb, Nn, 1024, 256, 1024, 1024, 1);
    k_gemm<<<dim3(256 / 128, Nn / 128), blk, 0, stream>>>(
        hidb, W2b, b2, ffn, (bf16*)nullptr, Nn, 256, 1024, 256, 256, 0);
    k_final<<<Nn, blk, 0, stream>>>(comb, ffn, ga3, bb3, (float*)d_out);
}
